// GraphSAGE_17428977287557
// MI455X (gfx1250) — matrix operand order repaired and checked by reading, not compiled, not run on hardware
//
#include <hip/hip_runtime.h>
#include <hip/hip_bf16.h>
#include <math.h>

#define NN 50000
#define EE 800000

typedef __attribute__((ext_vector_type(16))) __bf16 v16bf;
typedef __attribute__((ext_vector_type(16))) unsigned short v16u;
typedef __attribute__((ext_vector_type(8)))  unsigned short u16x8;
typedef __attribute__((ext_vector_type(8)))  float v8f;

__device__ __forceinline__ unsigned short f2bf(float f) {
    unsigned u = __float_as_uint(f);
    unsigned r = u + 0x7fffu + ((u >> 16) & 1u);   // round-to-nearest-even
    return (unsigned short)(r >> 16);
}

__device__ __forceinline__ unsigned pack2(float a, float b) {
    return (unsigned)f2bf(a) | ((unsigned)f2bf(b) << 16);
}

// ---------------- degree / inv_deg ----------------
__global__ void deg_kernel(const int* __restrict__ dst, float* __restrict__ deg) {
    int e = blockIdx.x * blockDim.x + threadIdx.x;
    if (e < EE) atomicAdd(&deg[dst[e]], 1.0f);
}

__global__ void invdeg_kernel(const float* __restrict__ deg, float* __restrict__ inv) {
    int i = blockIdx.x * blockDim.x + threadIdx.x;
    if (i < NN) inv[i] = 1.0f / fmaxf(deg[i], 1.0f);
}

// ---------------- weight pre-convert: f32 [K=128][N] -> bf16 transposed [N][128] ----------------
__global__ void convert_wt(const float* __restrict__ W, unsigned short* __restrict__ Wt, int N) {
    int i = blockIdx.x * blockDim.x + threadIdx.x;   // index into Wt (N*128)
    if (i < N * 128) {
        int n = i >> 7, k = i & 127;
        Wt[i] = f2bf(W[(size_t)k * N + n]);
    }
}

// ---------------- edge scatter-add (mean-agg numerator) ----------------
// 32 threads per edge, each handles 4 contiguous floats (float4 gather).
__global__ void scatter_kernel(const float* __restrict__ h,
                               const int* __restrict__ src,
                               const int* __restrict__ dst,
                               float* __restrict__ msg) {
    long long t = (long long)blockIdx.x * blockDim.x + threadIdx.x;
    int e = (int)(t >> 5);
    if (e >= EE) return;
    int ch = (int)(t & 31);
    int s = src[e], d = dst[e];
    const float4 v = *(const float4*)(h + (size_t)s * 128 + ch * 4);
    float* m = msg + (size_t)d * 128 + ch * 4;
    atomicAdd(m + 0, v.x);
    atomicAdd(m + 1, v.y);
    atomicAdd(m + 2, v.z);
    atomicAdd(m + 3, v.w);
}

// ---------------- fused SAGE layer: WMMA GEMM + bias (+ReLU + l2norm) ----------------
// Block: 16 output rows, NW = DOUT/16 waves (1 wave -> one 16x16 output tile).
// A tiles (self / neigh) staged as bf16 in LDS; B fragments read directly from
// pre-transposed bf16 weights in global (L2-resident), 2x global_load_b128 each.
template <int DOUT, bool LAST>
__global__ void sage_layer(const float* __restrict__ hself,
                           const float* __restrict__ msg,
                           const float* __restrict__ invdeg,
                           const unsigned short* __restrict__ WtS,  // [DOUT][128] bf16
                           const unsigned short* __restrict__ WtN,  // [DOUT][128] bf16
                           const float* __restrict__ bias,          // [DOUT]
                           float* __restrict__ out) {
    constexpr int NW = DOUT / 16;
    constexpr int NT = NW * 32;

    __shared__ __attribute__((aligned(16))) unsigned short As[16 * 128];
    __shared__ __attribute__((aligned(16))) unsigned short An[16 * 128];
    __shared__ float tile[16 * DOUT];
    __shared__ float rscale[16];

    const int row0 = blockIdx.x * 16;
    const int tid = threadIdx.x;
    const int lane = tid & 31;
    const int wave = tid >> 5;
    const int cN = lane & 15;     // N within tile (B/C/D) ; M for A
    const int hi = lane >> 4;     // lane half
    const int n0 = wave * 16;

    // Stage A tiles (16 rows x 128 K) as bf16; vectorized f32x4 -> u16x4 packs.
    for (int i = tid * 4; i < 16 * 128; i += NT * 4) {
        const float4 vs = *(const float4*)(hself + (size_t)row0 * 128 + i);
        const float4 vn = *(const float4*)(msg + (size_t)row0 * 128 + i);
        const float id = invdeg[row0 + (i >> 7)];
        uint2 ps, pn;
        ps.x = pack2(vs.x, vs.y);          ps.y = pack2(vs.z, vs.w);
        pn.x = pack2(vn.x * id, vn.y * id); pn.y = pack2(vn.z * id, vn.w * id);
        *(uint2*)(As + i) = ps;
        *(uint2*)(An + i) = pn;
    }
    __syncthreads();

    const u16x8* aSb = (const u16x8*)(As + cN * 128);
    const u16x8* aNb = (const u16x8*)(An + cN * 128);
    const u16x8* bSb = (const u16x8*)(WtS + (size_t)(n0 + cN) * 128);
    const u16x8* bNb = (const u16x8*)(WtN + (size_t)(n0 + cN) * 128);

    v8f acc = {};
#pragma unroll
    for (int kc = 0; kc < 4; ++kc) {
        // A fragment halves: K = kc*32 + hi*8 + [0..7]  and  + 16 + hi*8 + [0..7]
        u16x8 aS0 = aSb[kc * 4 + hi];
        u16x8 aS1 = aSb[kc * 4 + 2 + hi];
        u16x8 aN0 = aNb[kc * 4 + hi];
        u16x8 aN1 = aNb[kc * 4 + 2 + hi];
        // B fragment halves: K = kc*32 + hi*8 + [0..7]  and  + 16 + hi*8 + [0..7]
        u16x8 bS0 = bSb[kc * 4 + hi];
        u16x8 bS1 = bSb[kc * 4 + 2 + hi];
        u16x8 bN0 = bNb[kc * 4 + hi];
        u16x8 bN1 = bNb[kc * 4 + 2 + hi];

        v16u auS = __builtin_shufflevector(aS0, aS1, 0,1,2,3,4,5,6,7,8,9,10,11,12,13,14,15);
        v16u buS = __builtin_shufflevector(bS0, bS1, 0,1,2,3,4,5,6,7,8,9,10,11,12,13,14,15);
        v16u auN = __builtin_shufflevector(aN0, aN1, 0,1,2,3,4,5,6,7,8,9,10,11,12,13,14,15);
        v16u buN = __builtin_shufflevector(bN0, bN1, 0,1,2,3,4,5,6,7,8,9,10,11,12,13,14,15);

        acc = __builtin_amdgcn_wmma_f32_16x16x32_bf16(
            false, __builtin_bit_cast(v16bf, auS), false, __builtin_bit_cast(v16bf, buS),
            (short)0, acc, false, false);
        acc = __builtin_amdgcn_wmma_f32_16x16x32_bf16(
            false, __builtin_bit_cast(v16bf, auN), false, __builtin_bit_cast(v16bf, buN),
            (short)0, acc, false, false);
    }

    const float bval = bias[n0 + cN];

    if (LAST) {
#pragma unroll
        for (int v = 0; v < 8; ++v) {
            int r = v + 8 * hi;   // C/D: VGPR v -> M=v (lanes 0-15) / M=v+8 (lanes 16-31)
            out[(size_t)(row0 + r) * DOUT + n0 + cN] = acc[v] + bval;
        }
    } else {
#pragma unroll
        for (int v = 0; v < 8; ++v) {
            int r = v + 8 * hi;
            tile[r * DOUT + n0 + cN] = fmaxf(acc[v] + bval, 0.0f);
        }
        __syncthreads();
        if (tid < 16) {
            float s = 0.0f;
            for (int k = 0; k < DOUT; ++k) {
                float x = tile[tid * DOUT + k];
                s += x * x;
            }
            rscale[tid] = 1.0f / fmaxf(sqrtf(s), 1e-12f);
        }
        __syncthreads();
        for (int i = tid; i < 16 * DOUT; i += NT) {
            out[(size_t)row0 * DOUT + i] = tile[i] * rscale[i / DOUT];
        }
    }
}

extern "C" void kernel_launch(void* const* d_in, const int* in_sizes, int n_in,
                              void* d_out, int out_size, void* d_ws, size_t ws_size,
                              hipStream_t stream) {
    const float* feat = (const float*)d_in[0];
    const int* src = (const int*)d_in[1];
    const int* dst = (const int*)d_in[2];
    const float* Ws0 = (const float*)d_in[3];
    const float* Wn0 = (const float*)d_in[4];
    const float* b0 = (const float*)d_in[5];
    const float* Ws1 = (const float*)d_in[6];
    const float* Wn1 = (const float*)d_in[7];
    const float* b1 = (const float*)d_in[8];
    const float* Ws2 = (const float*)d_in[9];
    const float* Wn2 = (const float*)d_in[10];
    const float* b2 = (const float*)d_in[11];

    char* ws = (char*)d_ws;
    float* deg = (float*)ws;                 ws += (size_t)NN * sizeof(float);
    float* inv = (float*)ws;                 ws += (size_t)NN * sizeof(float);
    float* msg = (float*)ws;                 ws += (size_t)NN * 128 * sizeof(float);
    float* h1  = (float*)ws;                 ws += (size_t)NN * 128 * sizeof(float);
    float* h2  = (float*)ws;                 ws += (size_t)NN * 128 * sizeof(float);
    unsigned short* WtS0 = (unsigned short*)ws; ws += (size_t)128 * 128 * 2;
    unsigned short* WtN0 = (unsigned short*)ws; ws += (size_t)128 * 128 * 2;
    unsigned short* WtS1 = (unsigned short*)ws; ws += (size_t)128 * 128 * 2;
    unsigned short* WtN1 = (unsigned short*)ws; ws += (size_t)128 * 128 * 2;
    unsigned short* WtS2 = (unsigned short*)ws; ws += (size_t)64 * 128 * 2;
    unsigned short* WtN2 = (unsigned short*)ws; ws += (size_t)64 * 128 * 2;

    const int TPB = 256;
    const int degBlocks = (EE + TPB - 1) / TPB;
    const int invBlocks = (NN + TPB - 1) / TPB;
    const long long scatThreads = (long long)EE * 32;
    const int scatBlocks = (int)((scatThreads + TPB - 1) / TPB);
    const int rowBlocks = NN / 16;  // 3125, exact
    const int cw128 = (128 * 128 + TPB - 1) / TPB;
    const int cw64 = (64 * 128 + TPB - 1) / TPB;

    // degree + inv_deg (shared by all layers)
    hipMemsetAsync(deg, 0, (size_t)NN * sizeof(float), stream);
    deg_kernel<<<degBlocks, TPB, 0, stream>>>(dst, deg);
    invdeg_kernel<<<invBlocks, TPB, 0, stream>>>(deg, inv);

    // one-shot weight convert+transpose to bf16
    convert_wt<<<cw128, TPB, 0, stream>>>(Ws0, WtS0, 128);
    convert_wt<<<cw128, TPB, 0, stream>>>(Wn0, WtN0, 128);
    convert_wt<<<cw128, TPB, 0, stream>>>(Ws1, WtS1, 128);
    convert_wt<<<cw128, TPB, 0, stream>>>(Wn1, WtN1, 128);
    convert_wt<<<cw64,  TPB, 0, stream>>>(Ws2, WtS2, 64);
    convert_wt<<<cw64,  TPB, 0, stream>>>(Wn2, WtN2, 64);

    // ---- layer 0 ----
    hipMemsetAsync(msg, 0, (size_t)NN * 128 * sizeof(float), stream);
    scatter_kernel<<<scatBlocks, TPB, 0, stream>>>(feat, src, dst, msg);
    sage_layer<128, false><<<rowBlocks, 256, 0, stream>>>(feat, msg, inv, WtS0, WtN0, b0, h1);

    // ---- layer 1 ----
    hipMemsetAsync(msg, 0, (size_t)NN * 128 * sizeof(float), stream);
    scatter_kernel<<<scatBlocks, TPB, 0, stream>>>(h1, src, dst, msg);
    sage_layer<128, false><<<rowBlocks, 256, 0, stream>>>(h1, msg, inv, WtS1, WtN1, b1, h2);

    // ---- layer 2 (64-wide, no relu/norm) ----
    hipMemsetAsync(msg, 0, (size_t)NN * 128 * sizeof(float), stream);
    scatter_kernel<<<scatBlocks, TPB, 0, stream>>>(h2, src, dst, msg);
    sage_layer<64, true><<<rowBlocks, 128, 0, stream>>>(h2, msg, inv, WtS2, WtN2, b2, (float*)d_out);
}
